// MixedSparseAttention_42717744726490
// MI455X (gfx1250) — compile-verified, matrix operand order checked by reading, not run on hardware
//
#include <hip/hip_runtime.h>
#include <hip/hip_bf16.h>
#include <math.h>

// ---------------- problem constants (B=1) ----------------
#define N_TOK  4096
#define DIM    512
#define NHEADS 4          // per branch
#define HD     64
#define LDQKV  768        // qkv buffer row stride: [3][4][64]
#define SCALEF 0.125f     // 64^-0.5
#define TOPK   819        // int(4096*0.2)

typedef __attribute__((ext_vector_type(16))) __bf16 bf16x16;
typedef __attribute__((ext_vector_type(8)))  __bf16 bf16x8;
typedef __attribute__((ext_vector_type(8)))  float  f32x8;
typedef unsigned int u32x4 __attribute__((ext_vector_type(4)));
typedef int          i32x8 __attribute__((ext_vector_type(8)));
typedef int          i32x4 __attribute__((ext_vector_type(4)));

#if __has_builtin(__builtin_amdgcn_tensor_load_to_lds)
#define HAVE_TDM 1
#else
#define HAVE_TDM 0
#endif

__device__ __forceinline__ f32x8 wmma_bf16(bf16x16 a, bf16x16 b, f32x8 c) {
  // D = A(16x32 bf16) x B(32x16 bf16) + C(16x16 f32)
  return __builtin_amdgcn_wmma_f32_16x16x32_bf16(false, a, false, b, (short)0, c,
                                                 false, false);
}

#if HAVE_TDM
// Issue a TDM 2D tile copy: tile_d0 x tile_d1 elements of 2-byte data from a
// row-major tensor (dim0 stride = stride_d0 elements) into LDS at lds_off.
// D# layout per CDNA5 ISA.
__device__ __forceinline__ void tdm_load_2d_bf16(const __bf16* gsrc, unsigned lds_off,
                                                 int tensor_d0, int tensor_d1,
                                                 int tile_d0, int tile_d1, int stride_d0) {
  unsigned long long ga = (unsigned long long)(const void*)gsrc;
  u32x4 g0;
  g0[0] = 1u;                                          // count=1, user mode, no gather
  g0[1] = lds_off;                                     // [63:32] lds_addr (bytes)
  g0[2] = (unsigned)(ga & 0xFFFFFFFFu);                // [95:64] global_addr lo
  g0[3] = (unsigned)((ga >> 32) & 0x01FFFFFFu)         // [120:96] global_addr hi
        | 0x80000000u;                                 // [127:126] type=2 ("image")
  i32x8 g1;
  g1[0] = (1 << 16);                                   // data_size=1 -> 2 bytes
  g1[1] = (tensor_d0 & 0xFFFF) << 16;                  // [63:48] tensor_dim0 lo
  g1[2] = ((tensor_d0 >> 16) & 0xFFFF) |               // [79:64] tensor_dim0 hi
          ((tensor_d1 & 0xFFFF) << 16);                // [95:80] tensor_dim1 lo
  g1[3] = ((tensor_d1 >> 16) & 0xFFFF) |               // [111:96] ... hi
          ((tile_d0 & 0xFFFF) << 16);                  // tile_dim0
  g1[4] = (tile_d1 & 0xFFFF);                          // tile_dim1 (tile_dim2 = 0)
  g1[5] = stride_d0;                                   // tensor_dim0_stride lo32
  g1[6] = 0;                                           // stride hi / dim1_stride
  g1[7] = 0;
  i32x4 z4 = {0, 0, 0, 0};
#if defined(__clang_major__) && __clang_major__ >= 23
  i32x8 z8 = {0, 0, 0, 0, 0, 0, 0, 0};
  __builtin_amdgcn_tensor_load_to_lds(g0, g1, z4, z4, z8, 0);
#else
  __builtin_amdgcn_tensor_load_to_lds(g0, g1, z4, z4, 0);
#endif
}
#endif

// A-fragment (16x32, 16-bit): lane holds row `row`; element j -> K = k0 + (j/8)*16 + (lane>>4)*8 + j%8.
// The two 8-element runs are contiguous in memory -> two b128 loads.
__device__ __forceinline__ bf16x16 load_frag_row(const __bf16* base, int ld, int row, int k0) {
  int kb = ((threadIdx.x & 31) >> 4) * 8;
  const __bf16* p = base + (size_t)row * ld + k0 + kb;
  bf16x8 lo = *(const bf16x8*)p;
  bf16x8 hi = *(const bf16x8*)(p + 16);
  bf16x16 f;
#pragma unroll
  for (int j = 0; j < 8; ++j) { f[j] = lo[j]; f[8 + j] = hi[j]; }
  return f;
}

// B-fragment for row-major K x N matrix in global memory (fallback path for V):
// lane holds column n0+(lane&15), same K packing as A (strided loads).
__device__ __forceinline__ bf16x16 load_frag_colmajor(const __bf16* B, int ldb, int k0, int n0) {
  int lane = threadIdx.x & 31;
  int n = n0 + (lane & 15);
  int kb = (lane >> 4) * 8;
  bf16x16 f;
#pragma unroll
  for (int j = 0; j < 8; ++j) {
    f[j]     = B[(size_t)(k0 + kb + j) * ldb + n];
    f[8 + j] = B[(size_t)(k0 + 16 + kb + j) * ldb + n];
  }
  return f;
}

// ---------------- generic bf16 GEMM: C = A(MxK) * B(KxN) + bias ----------------
// B panel (K x 16 columns) is staged into LDS once per block via the Tensor Data
// Mover (tensor_load_to_lds + s_wait_tensorcnt); inner loop reads B from LDS.
// flags: bit0 = exact GELU on output, bit1 = accumulate into Cf
__global__ void gemm_wmma(const __bf16* __restrict__ A, const __bf16* __restrict__ B,
                          const float* __restrict__ bias, float* __restrict__ Cf,
                          __bf16* __restrict__ Cb, int Kdim, int lda, int ldb, int ldc,
                          int flags) {
  __shared__ __bf16 ldsB[512 * 16];                     // max Kdim=512 -> 16KB
  int n0 = blockIdx.x * 16, m0 = blockIdx.y * 16;
  int lane = threadIdx.x & 31;

#if HAVE_TDM
  // DMA the K x 16 bf16 weight panel into LDS (row stride in LDS = 16 elements).
  tdm_load_2d_bf16(B + n0, (unsigned)(size_t)(void*)ldsB,
                   /*tensor_d0=*/ldb, /*tensor_d1=*/Kdim,
                   /*tile_d0=*/16, /*tile_d1=*/Kdim, /*stride_d0=*/ldb);
#if __has_builtin(__builtin_amdgcn_s_wait_tensorcnt)
  __builtin_amdgcn_s_wait_tensorcnt(0);
#else
  asm volatile("s_wait_tensorcnt 0x0" ::: "memory");
#endif
  asm volatile("" ::: "memory");                        // keep LDS reads below the wait
#else
  for (int idx = lane; idx < Kdim * 16; idx += 32) {
    int k = idx >> 4, n = idx & 15;
    ldsB[idx] = B[(size_t)k * ldb + n0 + n];
  }
  __syncthreads();
#endif

  f32x8 c = {};
  for (int k0 = 0; k0 < Kdim; k0 += 32) {
    bf16x16 a = load_frag_row(A, lda, m0 + (lane & 15), k0);
    int nn = lane & 15, kb = (lane >> 4) * 8;
    bf16x16 b;
#pragma unroll
    for (int j = 0; j < 8; ++j) {
      b[j]     = ldsB[(k0 + kb + j) * 16 + nn];
      b[8 + j] = ldsB[(k0 + 16 + kb + j) * 16 + nn];
    }
    c = wmma_bf16(a, b, c);
  }
  int col = n0 + (lane & 15);
  float bv = bias ? bias[col] : 0.f;
#pragma unroll
  for (int i = 0; i < 8; ++i) {
    int row = m0 + (lane >> 4) * 8 + i;
    float v = c[i] + bv;
    if (flags & 1) v = 0.5f * v * (1.f + erff(v * 0.70710678118654752f));
    if (Cf) {
      if (flags & 2) Cf[(size_t)row * ldc + col] += v;
      else           Cf[(size_t)row * ldc + col] = v;
    }
    if (Cb) Cb[(size_t)row * ldc + col] = (__bf16)v;
  }
}

// ---------------- f32 -> bf16 convert ----------------
__global__ void cvt_bf16(const float* __restrict__ s, __bf16* __restrict__ d, int n) {
  int i = blockIdx.x * blockDim.x + threadIdx.x;
  if (i < n) d[i] = (__bf16)s[i];
}

// ---------------- saliency score GEMV: scores = h @ W_s2 + b_s2 ----------------
__global__ void saliency_kernel(const float* __restrict__ Hs, const float* __restrict__ W2,
                                const float* __restrict__ b2, float* __restrict__ scores) {
  int m = blockIdx.x * blockDim.x + threadIdx.x;
  if (m < N_TOK) {
    float s = b2[0];
    for (int k = 0; k < DIM / 2; ++k) s += Hs[(size_t)m * (DIM / 2) + k] * W2[k];
    scores[m] = s;
  }
}

// ---------------- exact top-K via single-block bitonic sort (score desc, idx asc) ----
__global__ void topk_mask_kernel(const float* __restrict__ scores, int* __restrict__ mask) {
  __shared__ float sv[N_TOK];
  __shared__ int   si[N_TOK];
  int t = threadIdx.x;
  for (int i = t; i < N_TOK; i += blockDim.x) { sv[i] = scores[i]; si[i] = i; mask[i] = 0; }
  __syncthreads();
  for (int k = 2; k <= N_TOK; k <<= 1) {
    for (int j = k >> 1; j > 0; j >>= 1) {
      for (int i = t; i < N_TOK; i += blockDim.x) {
        int ixj = i ^ j;
        if (ixj > i) {
          bool desc = ((i & k) == 0);
          float va = sv[i], vb = sv[ixj];
          int   ia = si[i], ib = si[ixj];
          bool a_first = (va > vb) || (va == vb && ia < ib);   // ranks earlier, descending
          bool swap = desc ? !a_first : a_first;
          if (swap) { sv[i] = vb; sv[ixj] = va; si[i] = ib; si[ixj] = ia; }
        }
      }
      __syncthreads();
    }
  }
  for (int i = t; i < TOPK; i += blockDim.x) mask[si[i]] = 1;
}

// ---------------- attention pass 1: per-row online (max, sumexp) -------------------
// grid = (N/16 qtiles, NHEADS); 1 wave/block. mask==nullptr for local branch.
__global__ void attn_stats(const __bf16* __restrict__ qkv, const int* __restrict__ mask,
                           float* __restrict__ mbuf, float* __restrict__ lbuf) {
  int head = blockIdx.y, q0 = blockIdx.x * 16;
  int lane = threadIdx.x & 31;
  const __bf16* Q  = qkv + head * HD;
  const __bf16* Km = qkv + (NHEADS + head) * HD;
  bf16x16 a0 = load_frag_row(Q, LDQKV, q0 + (lane & 15), 0);
  bf16x16 a1 = load_frag_row(Q, LDQKV, q0 + (lane & 15), 32);
  float mrun[8], lrun[8];
#pragma unroll
  for (int i = 0; i < 8; ++i) { mrun[i] = -INFINITY; lrun[i] = 0.f; }
  for (int kt = 0; kt < N_TOK; kt += 16) {
    int col = kt + (lane & 15);
    bool ok = mask ? (mask[col] != 0) : true;
    bf16x16 b0 = load_frag_row(Km, LDQKV, col, 0);    // K^T B-fragment (d runs contiguous)
    bf16x16 b1 = load_frag_row(Km, LDQKV, col, 32);
    if (kt + 16 < N_TOK)
      __builtin_prefetch(Km + (size_t)(col + 16) * LDQKV, 0, 1);   // global_prefetch_b8
    f32x8 c = {};
    c = wmma_bf16(a0, b0, c);
    c = wmma_bf16(a1, b1, c);
#pragma unroll
    for (int i = 0; i < 8; ++i) {
      float v = ok ? c[i] * SCALEF : -INFINITY;
      float rm = v;
#pragma unroll
      for (int off = 1; off < 16; off <<= 1) rm = fmaxf(rm, __shfl_xor(rm, off, 32));
      float newm = fmaxf(mrun[i], rm);
      float se = (v > -1e30f) ? __expf(v - newm) : 0.f;     // branchless, EXEC stays full
#pragma unroll
      for (int off = 1; off < 16; off <<= 1) se += __shfl_xor(se, off, 32);
      float corr = (mrun[i] > -1e30f) ? __expf(mrun[i] - newm) : 0.f;
      lrun[i] = lrun[i] * corr + se;
      mrun[i] = newm;
    }
  }
  if ((lane & 15) == 0) {
#pragma unroll
    for (int i = 0; i < 8; ++i) {
      int row = q0 + (lane >> 4) * 8 + i;
      mbuf[head * N_TOK + row] = mrun[i];
      lbuf[head * N_TOK + row] = lrun[i];
    }
  }
}

// ---------------- attention pass 2: head-mean probabilities -> d_out ---------------
// grid = (N/16 ktiles, N/16 qtiles); loops heads inside -> single deterministic write.
__global__ void attn_probs(const __bf16* __restrict__ qkv, const int* __restrict__ mask,
                           const float* __restrict__ mbuf, const float* __restrict__ lbuf,
                           float* __restrict__ attn_out) {
  int q0 = blockIdx.y * 16, k0 = blockIdx.x * 16;
  int lane = threadIdx.x & 31;
  int col = k0 + (lane & 15);
  bool ok = mask ? (mask[col] != 0) : true;
  float acc[8] = {0.f, 0.f, 0.f, 0.f, 0.f, 0.f, 0.f, 0.f};
  for (int h = 0; h < NHEADS; ++h) {
    const __bf16* Q  = qkv + h * HD;
    const __bf16* Km = qkv + (NHEADS + h) * HD;
    bf16x16 a0 = load_frag_row(Q, LDQKV, q0 + (lane & 15), 0);
    bf16x16 a1 = load_frag_row(Q, LDQKV, q0 + (lane & 15), 32);
    bf16x16 b0 = load_frag_row(Km, LDQKV, col, 0);
    bf16x16 b1 = load_frag_row(Km, LDQKV, col, 32);
    f32x8 c = {};
    c = wmma_bf16(a0, b0, c);
    c = wmma_bf16(a1, b1, c);
#pragma unroll
    for (int i = 0; i < 8; ++i) {
      int row = q0 + (lane >> 4) * 8 + i;
      float p = ok ? __expf(c[i] * SCALEF - mbuf[h * N_TOK + row]) / lbuf[h * N_TOK + row] : 0.f;
      acc[i] += p;
    }
  }
#pragma unroll
  for (int i = 0; i < 8; ++i) {
    int row = q0 + (lane >> 4) * 8 + i;
    attn_out[(size_t)row * N_TOK + col] = acc[i] * 0.25f;   // coalesced b32 stores
  }
}

// ---------------- attention pass 3: O_h = P_h @ V_h ---------------------------------
// grid = (N/16 qtiles, NHEADS); 1 wave/block. P is recomputed from (m,l) stats and
// relaid out through LDS; V chunks (32x64 bf16) are double-buffered into LDS by the
// Tensor Data Mover, overlapping DMA with the softmax VALU + WMMA work.
__global__ void attn_av(const __bf16* __restrict__ qkv, const int* __restrict__ mask,
                        const float* __restrict__ mbuf, const float* __restrict__ lbuf,
                        __bf16* __restrict__ Obuf) {
  __shared__ float pl[16 * 32];
#if HAVE_TDM
  __shared__ __bf16 vbuf[2][32 * 64];                   // 2 x 4KB V chunk double buffer
#endif
  int head = blockIdx.y, q0 = blockIdx.x * 16;
  int lane = threadIdx.x & 31;
  const __bf16* Q  = qkv + head * HD;
  const __bf16* Km = qkv + (NHEADS + head) * HD;
  const __bf16* V  = qkv + (2 * NHEADS + head) * HD;
  bf16x16 a0 = load_frag_row(Q, LDQKV, q0 + (lane & 15), 0);
  bf16x16 a1 = load_frag_row(Q, LDQKV, q0 + (lane & 15), 32);
  float mh[8], lh[8];
#pragma unroll
  for (int i = 0; i < 8; ++i) {
    int row = q0 + (lane >> 4) * 8 + i;
    mh[i] = mbuf[head * N_TOK + row];
    lh[i] = lbuf[head * N_TOK + row];
  }
  f32x8 oc0 = {}, oc1 = {}, oc2 = {}, oc3 = {};

#if HAVE_TDM
  // prime the pipeline: DMA V rows [0,32) x 64 dims into buffer 0
  tdm_load_2d_bf16(V, (unsigned)(size_t)(void*)&vbuf[0][0],
                   /*tensor_d0=*/HD, /*tensor_d1=*/N_TOK,
                   /*tile_d0=*/HD, /*tile_d1=*/32, /*stride_d0=*/LDQKV);
#endif
  int cur = 0;
  for (int k0 = 0; k0 < N_TOK; k0 += 32) {
#if HAVE_TDM
    if (k0 + 32 < N_TOK)                                 // issue next chunk's DMA early
      tdm_load_2d_bf16(V + (size_t)(k0 + 32) * LDQKV,
                       (unsigned)(size_t)(void*)&vbuf[cur ^ 1][0],
                       HD, N_TOK, HD, 32, LDQKV);
#endif
#pragma unroll
    for (int sub = 0; sub < 2; ++sub) {
      int kc = k0 + sub * 16;
      int col = kc + (lane & 15);
      bool ok = mask ? (mask[col] != 0) : true;
      bf16x16 b0 = load_frag_row(Km, LDQKV, col, 0);
      bf16x16 b1 = load_frag_row(Km, LDQKV, col, 32);
      f32x8 c = {};
      c = wmma_bf16(a0, b0, c);
      c = wmma_bf16(a1, b1, c);
#pragma unroll
      for (int i = 0; i < 8; ++i) {
        float p = ok ? __expf(c[i] * SCALEF - mh[i]) / lh[i] : 0.f;
        pl[((lane >> 4) * 8 + i) * 32 + sub * 16 + (lane & 15)] = p;  // C-layout scatter
      }
    }
    if (k0 + 32 < N_TOK)
      __builtin_prefetch(Km + (size_t)(k0 + 32 + (lane & 15)) * LDQKV, 0, 1);
    __syncthreads();
    bf16x16 pa;                               // re-read P as A-fragment
    {
      int m = lane & 15, kb = (lane >> 4) * 8;
      const float* p = &pl[m * 32 + kb];
#pragma unroll
      for (int j = 0; j < 8; ++j) { pa[j] = (__bf16)p[j]; pa[8 + j] = (__bf16)p[16 + j]; }
    }
    __syncthreads();

#if HAVE_TDM
    // TENSORcnt retires in order: after issuing chunk i+1, waiting <=1 guarantees
    // chunk i (in vbuf[cur]) has landed; last iteration waits for 0 outstanding.
#if __has_builtin(__builtin_amdgcn_s_wait_tensorcnt)
    if (k0 + 32 < N_TOK) __builtin_amdgcn_s_wait_tensorcnt(1);
    else                 __builtin_amdgcn_s_wait_tensorcnt(0);
#else
    if (k0 + 32 < N_TOK) asm volatile("s_wait_tensorcnt 0x1" ::: "memory");
    else                 asm volatile("s_wait_tensorcnt 0x0" ::: "memory");
#endif
    asm volatile("" ::: "memory");
    {
      const __bf16* vb = &vbuf[cur][0];                 // [kk][d], row stride 64
      int kb = (lane >> 4) * 8;
#pragma unroll
      for (int nt = 0; nt < 4; ++nt) {
        int n = nt * 16 + (lane & 15);
        bf16x16 f;
#pragma unroll
        for (int j = 0; j < 8; ++j) {
          f[j]     = vb[(kb + j) * HD + n];
          f[8 + j] = vb[(16 + kb + j) * HD + n];
        }
        if (nt == 0) oc0 = wmma_bf16(pa, f, oc0);
        else if (nt == 1) oc1 = wmma_bf16(pa, f, oc1);
        else if (nt == 2) oc2 = wmma_bf16(pa, f, oc2);
        else oc3 = wmma_bf16(pa, f, oc3);
      }
    }
#else
    oc0 = wmma_bf16(pa, load_frag_colmajor(V, LDQKV, k0, 0),  oc0);
    oc1 = wmma_bf16(pa, load_frag_colmajor(V, LDQKV, k0, 16), oc1);
    oc2 = wmma_bf16(pa, load_frag_colmajor(V, LDQKV, k0, 32), oc2);
    oc3 = wmma_bf16(pa, load_frag_colmajor(V, LDQKV, k0, 48), oc3);
#endif
    cur ^= 1;
  }
#pragma unroll
  for (int i = 0; i < 8; ++i) {
    int row = q0 + (lane >> 4) * 8 + i;
    __bf16* o = Obuf + (size_t)row * (NHEADS * HD) + head * HD + (lane & 15);
    o[0] = (__bf16)oc0[i]; o[16] = (__bf16)oc1[i]; o[32] = (__bf16)oc2[i]; o[48] = (__bf16)oc3[i];
  }
}

// =======================================================================================
extern "C" void kernel_launch(void* const* d_in, const int* in_sizes, int n_in,
                              void* d_out, int out_size, void* d_ws, size_t ws_size,
                              hipStream_t stream) {
  (void)in_sizes; (void)n_in; (void)out_size; (void)ws_size;
  const float* x       = (const float*)d_in[0];
  const float* W_lqkv  = (const float*)d_in[1];
  const float* b_lqkv  = (const float*)d_in[2];
  const float* W_cqkv  = (const float*)d_in[3];
  const float* b_cqkv  = (const float*)d_in[4];
  const float* W_lproj = (const float*)d_in[5];
  const float* b_lproj = (const float*)d_in[6];
  const float* W_cproj = (const float*)d_in[7];
  const float* b_cproj = (const float*)d_in[8];
  const float* W_s1    = (const float*)d_in[9];
  const float* b_s1    = (const float*)d_in[10];
  const float* W_s2    = (const float*)d_in[11];
  const float* b_s2    = (const float*)d_in[12];

  // ---- workspace carve-out (~28 MB) ----
  char* w = (char*)d_ws;
  auto carve = [&](size_t bytes) { char* p = w; w += (bytes + 255) & ~(size_t)255; return p; };
  __bf16* xb      = (__bf16*)carve((size_t)N_TOK * DIM * 2);
  __bf16* Wlqkvb  = (__bf16*)carve((size_t)DIM * 768 * 2);
  __bf16* Wcqkvb  = (__bf16*)carve((size_t)DIM * 768 * 2);
  __bf16* Ws1b    = (__bf16*)carve((size_t)DIM * 256 * 2);
  __bf16* Wlprojb = (__bf16*)carve((size_t)256 * DIM * 2);
  __bf16* Wcprojb = (__bf16*)carve((size_t)256 * DIM * 2);
  __bf16* lqkvb   = (__bf16*)carve((size_t)N_TOK * 768 * 2);
  __bf16* cqkvb   = (__bf16*)carve((size_t)N_TOK * 768 * 2);
  float*  salh    = (float*)carve((size_t)N_TOK * 256 * 4);
  float*  scores  = (float*)carve((size_t)N_TOK * 4);
  int*    maskbuf = (int*)carve((size_t)N_TOK * 4);
  float*  m_l     = (float*)carve((size_t)NHEADS * N_TOK * 4);
  float*  l_l     = (float*)carve((size_t)NHEADS * N_TOK * 4);
  float*  m_c     = (float*)carve((size_t)NHEADS * N_TOK * 4);
  float*  l_c     = (float*)carve((size_t)NHEADS * N_TOK * 4);
  __bf16* LOb     = (__bf16*)carve((size_t)N_TOK * 256 * 2);
  __bf16* COb     = (__bf16*)carve((size_t)N_TOK * 256 * 2);

  float* out   = (float*)d_out;                        // (4096, 512)
  float* attnL = out + (size_t)N_TOK * DIM;            // (4096, 4096) local head-mean
  float* attnC = attnL + (size_t)N_TOK * N_TOK;        // (4096, 4096) content head-mean

  // ---- 1) f32 -> bf16 conversions ----
  auto cvt = [&](const float* s, __bf16* d, int n) {
    cvt_bf16<<<dim3((n + 255) / 256), dim3(256), 0, stream>>>(s, d, n);
  };
  cvt(x, xb, N_TOK * DIM);
  cvt(W_lqkv, Wlqkvb, DIM * 768);
  cvt(W_cqkv, Wcqkvb, DIM * 768);
  cvt(W_s1, Ws1b, DIM * 256);
  cvt(W_lproj, Wlprojb, 256 * DIM);
  cvt(W_cproj, Wcprojb, 256 * DIM);

  // ---- 2) QKV + saliency GEMMs (WMMA bf16, TDM-staged B panels) ----
  gemm_wmma<<<dim3(768 / 16, N_TOK / 16), dim3(32), 0, stream>>>(
      xb, Wlqkvb, b_lqkv, nullptr, lqkvb, DIM, DIM, 768, 768, 0);
  gemm_wmma<<<dim3(768 / 16, N_TOK / 16), dim3(32), 0, stream>>>(
      xb, Wcqkvb, b_cqkv, nullptr, cqkvb, DIM, DIM, 768, 768, 0);
  gemm_wmma<<<dim3(256 / 16, N_TOK / 16), dim3(32), 0, stream>>>(
      xb, Ws1b, b_s1, salh, nullptr, DIM, DIM, 256, 256, /*gelu*/ 1);

  // ---- 3) saliency scores + exact top-K mask ----
  saliency_kernel<<<dim3((N_TOK + 255) / 256), dim3(256), 0, stream>>>(salh, W_s2, b_s2, scores);
  topk_mask_kernel<<<dim3(1), dim3(1024), 0, stream>>>(scores, maskbuf);

  // ---- 4) attention: stats -> head-mean probs (to d_out) -> P@V ----
  attn_stats<<<dim3(N_TOK / 16, NHEADS), dim3(32), 0, stream>>>(lqkvb, nullptr, m_l, l_l);
  attn_stats<<<dim3(N_TOK / 16, NHEADS), dim3(32), 0, stream>>>(cqkvb, maskbuf, m_c, l_c);

  attn_probs<<<dim3(N_TOK / 16, N_TOK / 16), dim3(32), 0, stream>>>(lqkvb, nullptr, m_l, l_l, attnL);
  attn_probs<<<dim3(N_TOK / 16, N_TOK / 16), dim3(32), 0, stream>>>(cqkvb, maskbuf, m_c, l_c, attnC);

  attn_av<<<dim3(N_TOK / 16, NHEADS), dim3(32), 0, stream>>>(lqkvb, nullptr, m_l, l_l, LOb);
  attn_av<<<dim3(N_TOK / 16, NHEADS), dim3(32), 0, stream>>>(cqkvb, maskbuf, m_c, l_c, COb);

  // ---- 5) output projections: out = LO@W_lproj + b_lproj + CO@W_cproj + b_cproj ----
  gemm_wmma<<<dim3(DIM / 16, N_TOK / 16), dim3(32), 0, stream>>>(
      LOb, Wlprojb, b_lproj, out, nullptr, 256, 256, DIM, DIM, 0);
  gemm_wmma<<<dim3(DIM / 16, N_TOK / 16), dim3(32), 0, stream>>>(
      COb, Wcprojb, b_cproj, out, nullptr, 256, 256, DIM, DIM, /*accumulate*/ 2);
}
